// WindowAttention_2482491097523
// MI455X (gfx1250) — matrix operand order repaired and checked by reading, not compiled, not run on hardware
//
#include <hip/hip_runtime.h>
#include <hip/hip_bf16.h>
#include <math.h>

// ---------- types for WMMA ----------
typedef __attribute__((ext_vector_type(16))) __bf16          v16bf;
typedef __attribute__((ext_vector_type(8)))  float           v8f;
typedef __attribute__((ext_vector_type(8)))  unsigned short  v8u;
typedef __attribute__((ext_vector_type(16))) unsigned short  v16u;

// fp32 -> bf16 (round to nearest even)
__device__ __forceinline__ unsigned short f2bf(float f) {
    union { float f; unsigned u; } v; v.f = f;
    unsigned r = v.u + 0x7FFFu + ((v.u >> 16) & 1u);
    return (unsigned short)(r >> 16);
}

// Build a 16-half fragment from two 16-byte chunks (ds_load_b128 / global_load_b128).
__device__ __forceinline__ v16bf make_frag(const unsigned short* p0, const unsigned short* p1) {
    v8u a = *reinterpret_cast<const v8u*>(p0);
    v8u b = *reinterpret_cast<const v8u*>(p1);
    v16u f;
#pragma unroll
    for (int j = 0; j < 8; ++j) { f[j] = a[j]; f[j + 8] = b[j]; }
    return __builtin_bit_cast(v16bf, f);
}

// A fragment (16x32, M x K), row-major source, ld in halfs.
// ISA layout: lane L holds row M=L%16; halves 0..7 -> K = (L/16)*8 + 0..7, halves 8..15 -> +16.
__device__ __forceinline__ v16bf load_frag_a(const unsigned short* base, int ld,
                                             int row0, int k0, int lane) {
    const unsigned short* p0 = base + (row0 + (lane & 15)) * ld + k0 + ((lane >> 4) << 3);
    return make_frag(p0, p0 + 16);
}

// B fragment (32x16, K x N) from an N-major (transposed) source, ldt in halfs.
// ISA layout: lane L holds col N=L%16; halves 0..7 -> K = (L/16)*8 + 0..7, halves 8..15 -> +16.
__device__ __forceinline__ v16bf load_frag_b(const unsigned short* baseT, int ldt,
                                             int n0, int k0, int lane) {
    const unsigned short* p = baseT + (n0 + (lane & 15)) * ldt + k0 + ((lane >> 4) << 3);
    return make_frag(p, p + 16);
}

__device__ __forceinline__ v8f wmma_bf16(v16bf a, v16bf b, v8f c) {
    return __builtin_amdgcn_wmma_f32_16x16x32_bf16(false, a, false, b, (short)0, c, false, false);
}

// ---------- workspace layout (bytes) ----------
// [0, 98304)        : WqkvT  bf16 [384][128]
// [98304, 131072)   : WprojT bf16 [128][128]
// [131072, 169488)  : rel_bias f32 [4][49][49]
#define WS_WQKVT   0
#define WS_WPROJT  98304
#define WS_RELBIAS 131072

// ---------- prep: transpose weights to bf16, precompute relative-position bias ----------
__global__ void swin_prep_kernel(const float* __restrict__ w_qkv,
                                 const float* __restrict__ w_proj,
                                 const float* __restrict__ bias_table,
                                 unsigned short* __restrict__ wqkvT,
                                 unsigned short* __restrict__ wprojT,
                                 float* __restrict__ rel_bias) {
    int i = blockIdx.x * blockDim.x + threadIdx.x;
    if (i < 49152) {                       // WqkvT[n][k] = w_qkv[k][n], (128 x 384) -> (384 x 128)
        int n = i >> 7, k = i & 127;
        wqkvT[i] = f2bf(w_qkv[k * 384 + n]);
    } else if (i < 49152 + 16384) {        // WprojT[n][k] = w_proj[k][n]
        int j = i - 49152;
        int n = j >> 7, k = j & 127;
        wprojT[j] = f2bf(w_proj[k * 128 + n]);
    } else if (i < 49152 + 16384 + 9604) { // rel_bias[h][m][n]
        int j = i - (49152 + 16384);
        int h = j / 2401, r = j % 2401;
        int m = r / 49, n = r % 49;
        int ym = m / 7, xm = m % 7, yn = n / 7, xn = n % 7;
        int idx = (ym - yn + 6) * 13 + (xm - xn + 6);
        rel_bias[j] = bias_table[idx * 4 + h];
    }
}

// ---------- fused window attention: one workgroup per window ----------
__global__ __launch_bounds__(256) void swin_attn_kernel(
    const float* __restrict__ x,            // (4096, 49, 128)
    const float* __restrict__ mask,         // (64, 49, 49)
    const float* __restrict__ b_qkv,        // (384,)
    const float* __restrict__ b_proj,       // (128,)
    const unsigned short* __restrict__ wqkvT,   // bf16 (384,128)
    const unsigned short* __restrict__ wprojT,  // bf16 (128,128)
    const float* __restrict__ rel_bias,     // (4,49,49)
    float* __restrict__ out)                // (4096, 49, 128)
{
    const int b    = blockIdx.x;
    const int tid  = threadIdx.x;
    const int wave = tid >> 5;
    const int lane = tid & 31;
    const int wi   = b & 63;            // mask window index = b % nW

    extern __shared__ char smem[];
    unsigned short* sX    = (unsigned short*)(smem);          // [64][128] bf16 (phase 1)
    float*          attnF = (float*)(smem);                   // [64][64]  f32  (aliases sX)
    unsigned short* sQ    = (unsigned short*)(smem + 16384);  // [4][64][32] bf16 (q*scale)
    unsigned short* sK    = (unsigned short*)(smem + 32768);  // [4][64][32] bf16
    unsigned short* sVT   = (unsigned short*)(smem + 49152);  // [4][32][64] bf16 (v transposed)
    unsigned short* attnB = (unsigned short*)(smem + 65536);  // [64][64] bf16 softmax probs
    unsigned short* sO    = (unsigned short*)(smem + 73728);  // [64][128] bf16 attention output

    // ---- stage X tile (49x128, zero-padded to 64) as bf16 ----
    const float* xb = x + (size_t)b * 49 * 128;
    for (int i = tid; i < 64 * 128; i += 256) {
        int m = i >> 7, c = i & 127;
        sX[i] = f2bf((m < 49) ? xb[m * 128 + c] : 0.0f);
    }
    __syncthreads();

    // ---- phase 1: QKV = X @ Wqkv + b ; 96 tiles of 16x16, K=128 ----
    const float scale = 0.17677669529663687f;   // 32^-0.5
    for (int t = wave; t < 96; t += 8) {
        int mi = t / 24, ni = t % 24;
        v8f acc = {};
#pragma unroll
        for (int ks = 0; ks < 4; ++ks) {
            v16bf fa = load_frag_a(sX, 128, mi * 16, ks * 32, lane);
            v16bf fb = load_frag_b(wqkvT, 128, ni * 16, ks * 32, lane);
            acc = wmma_bf16(fa, fb, acc);
        }
        int ng = ni * 16 + (lane & 15);
        float bias = b_qkv[ng];
        int dd = ng & 31;
#pragma unroll
        for (int r = 0; r < 8; ++r) {
            int m = mi * 16 + r + ((lane >> 4) << 3);
            float val = acc[r] + bias;
            if (ng < 128) {                 // Q (pre-scaled)
                int h = ng >> 5;
                sQ[h * 2048 + m * 32 + dd] = f2bf(val * scale);
            } else if (ng < 256) {          // K
                int h = (ng - 128) >> 5;
                sK[h * 2048 + m * 32 + dd] = f2bf(val);
            } else {                        // V, stored transposed [dd][m]
                int h = (ng - 256) >> 5;
                sVT[h * 2048 + dd * 64 + m] = f2bf(val);
            }
        }
    }
    __syncthreads();

    // ---- per head: scores -> softmax -> attn@V ----
    for (int h = 0; h < 4; ++h) {
        // scores: q(49x32) @ k^T -> 16 tiles, single WMMA each (K=32)
        for (int t = wave; t < 16; t += 8) {
            int mi = t >> 2, ni = t & 3;
            v16bf fa = load_frag_a(sQ + h * 2048, 32, mi * 16, 0, lane);
            v16bf fb = load_frag_b(sK + h * 2048, 32, ni * 16, 0, lane);  // k row-major == N-major B
            v8f acc = {};
            acc = wmma_bf16(fa, fb, acc);
            int n = ni * 16 + (lane & 15);
#pragma unroll
            for (int r = 0; r < 8; ++r) {
                int m = mi * 16 + r + ((lane >> 4) << 3);
                float val;
                if (n < 49) {
                    float add = 0.0f;
                    if (m < 49)
                        add = rel_bias[h * 2401 + m * 49 + n] + mask[wi * 2401 + m * 49 + n];
                    val = acc[r] + add;
                } else {
                    val = -1e30f;           // padded keys vanish under softmax
                }
                attnF[m * 64 + n] = val;
            }
        }
        __syncthreads();

        // softmax over rows (49 valid rows; pad rows zeroed so K-padding is clean)
        if (tid < 64) {
            if (tid < 49) {
                const float* row = attnF + tid * 64;
                float mx = -1e30f;
                for (int j = 0; j < 49; ++j) mx = fmaxf(mx, row[j]);
                float s = 0.0f;
                for (int j = 0; j < 49; ++j) s += __expf(row[j] - mx);
                float rinv = 1.0f / s;
                for (int j = 0; j < 49; ++j) attnB[tid * 64 + j] = f2bf(__expf(row[j] - mx) * rinv);
                for (int j = 49; j < 64; ++j) attnB[tid * 64 + j] = 0;
            } else {
                for (int j = 0; j < 64; ++j) attnB[tid * 64 + j] = 0;
            }
        }
        __syncthreads();

        // out_h = attn(64x64) @ v(64x32): 8 tiles, one per wave, K=64 (2 WMMAs)
        {
            int mi = wave >> 1, di = wave & 1;
            v8f acc = {};
#pragma unroll
            for (int kk = 0; kk < 2; ++kk) {
                v16bf fa = load_frag_a(attnB, 64, mi * 16, kk * 32, lane);
                v16bf fb = load_frag_b(sVT + h * 2048, 64, di * 16, kk * 32, lane);
                acc = wmma_bf16(fa, fb, acc);
            }
            int n = h * 32 + di * 16 + (lane & 15);
#pragma unroll
            for (int r = 0; r < 8; ++r) {
                int m = mi * 16 + r + ((lane >> 4) << 3);
                sO[m * 128 + n] = f2bf(acc[r]);
            }
        }
        __syncthreads();
    }

    // ---- phase 4: out = sO @ Wproj + b_proj ; 32 tiles, K=128 ----
    float* ob = out + (size_t)b * 49 * 128;
    for (int t = wave; t < 32; t += 8) {
        int mi = t >> 3, ni = t & 7;
        v8f acc = {};
#pragma unroll
        for (int ks = 0; ks < 4; ++ks) {
            v16bf fa = load_frag_a(sO, 128, mi * 16, ks * 32, lane);
            v16bf fb = load_frag_b(wprojT, 128, ni * 16, ks * 32, lane);
            acc = wmma_bf16(fa, fb, acc);
        }
        int n = ni * 16 + (lane & 15);
        float bias = b_proj[n];
#pragma unroll
        for (int r = 0; r < 8; ++r) {
            int m = mi * 16 + r + ((lane >> 4) << 3);
            if (m < 49) ob[m * 128 + n] = acc[r] + bias;
        }
    }
}

extern "C" void kernel_launch(void* const* d_in, const int* in_sizes, int n_in,
                              void* d_out, int out_size, void* d_ws, size_t ws_size,
                              hipStream_t stream) {
    const float* x          = (const float*)d_in[0];
    const float* mask       = (const float*)d_in[1];
    const float* w_qkv      = (const float*)d_in[2];
    const float* b_qkv      = (const float*)d_in[3];
    const float* bias_table = (const float*)d_in[4];
    const float* w_proj     = (const float*)d_in[5];
    const float* b_proj     = (const float*)d_in[6];
    float* out = (float*)d_out;

    unsigned short* wqkvT  = (unsigned short*)((char*)d_ws + WS_WQKVT);
    unsigned short* wprojT = (unsigned short*)((char*)d_ws + WS_WPROJT);
    float*          relb   = (float*)((char*)d_ws + WS_RELBIAS);

    // allow >64KB dynamic LDS (gfx1250 WGP has 320KB)
    static const int kDynLds = 90112;
    hipFuncSetAttribute((const void*)swin_attn_kernel,
                        hipFuncAttributeMaxDynamicSharedMemorySize, kDynLds);

    // prep: weight transpose + bf16 + relative-position bias table
    swin_prep_kernel<<<(49152 + 16384 + 9604 + 255) / 256, 256, 0, stream>>>(
        w_qkv, w_proj, bias_table, wqkvT, wprojT, relb);

    // fused attention: one block per window
    swin_attn_kernel<<<4096, 256, kDynLds, stream>>>(
        x, mask, b_qkv, b_proj, wqkvT, wprojT, relb, out);
}
